// GlobalRelationalModel_74242804678696
// MI455X (gfx1250) — matrix operand order repaired and checked by reading, not compiled, not run on hardware
//
#include <hip/hip_runtime.h>
#include <cmath>

// ---------------- types ----------------
typedef __attribute__((ext_vector_type(16))) __bf16 v16bf;
typedef __attribute__((ext_vector_type(8)))  float  v8f;

union Frag16 { v16bf v; int4 q[2]; };

// A fragment (16x32, 16-bit): lane m (0-15) rows, lane-half selects K chunk:
// lanes 0-15: K = {0..7, 16..23}, lanes 16-31: K = {8..15, 24..31}
__device__ inline v16bf load_fragA(const __bf16* A, int ld, int m0, int k0, int lane) {
    int r = m0 + (lane & 15);
    int h = (lane >> 4) << 3;               // 0 or 8
    const __bf16* p = A + (size_t)r * ld + k0 + h;
    Frag16 f;
    f.q[0] = *(const int4*)p;               // K = k0+h .. +7
    f.q[1] = *(const int4*)(p + 16);        // K = k0+16+h .. +7
    return f.v;
}

// B fragment (32x16, 16-bit): lane n (0-15) columns; lanes 0-15 hold K {0..7, 16..23},
// lanes 16-31 hold K {8..15, 24..31} (same as A). W is [N,K] row-major,
// so "column n of B" == row n of W -> contiguous load.
__device__ inline v16bf load_fragB(const __bf16* W, int ld, int n0, int k0, int lane) {
    int r = n0 + (lane & 15);
    int h = (lane >> 4) << 3;               // 0 or 8
    const __bf16* p = W + (size_t)r * ld + k0 + h;
    Frag16 f;
    f.q[0] = *(const int4*)p;
    f.q[1] = *(const int4*)(p + 16);
    return f.v;
}

// ---------------- generic WMMA GEMM:  out[m,n] = act(scale[n]*(A W^T)[m,n] + bias[n]) (+resid)
// Requirements: M % 128 == 0 (grid.y = M/128), N % 64 == 0 (grid.x = N/64), Kd % 32 == 0.
// Each wave computes a 16x64 strip: A-fragment reused across 4 B tiles.
__global__ __launch_bounds__(256)
void gemm_bf16(const __bf16* __restrict__ A, const __bf16* __restrict__ W,
               const float* __restrict__ scale, const float* __restrict__ bias,
               const float* __restrict__ resid, float* __restrict__ outF,
               __bf16* __restrict__ outB, int M, int N, int Kd, int ldOut, int relu)
{
    int lane = threadIdx.x & 31;
    int wv   = threadIdx.x >> 5;
    int m0 = blockIdx.y * 128 + wv * 16;
    int nb = blockIdx.x * 64;

    v8f acc0 = {}, acc1 = {}, acc2 = {}, acc3 = {};
    for (int k0 = 0; k0 < Kd; k0 += 32) {
        v16bf a  = load_fragA(A, Kd, m0, k0, lane);
        v16bf b0 = load_fragB(W, Kd, nb,      k0, lane);
        v16bf b1 = load_fragB(W, Kd, nb + 16, k0, lane);
        v16bf b2 = load_fragB(W, Kd, nb + 32, k0, lane);
        v16bf b3 = load_fragB(W, Kd, nb + 48, k0, lane);
        acc0 = __builtin_amdgcn_wmma_f32_16x16x32_bf16(false, a, false, b0, (short)0, acc0, false, false);
        acc1 = __builtin_amdgcn_wmma_f32_16x16x32_bf16(false, a, false, b1, (short)0, acc1, false, false);
        acc2 = __builtin_amdgcn_wmma_f32_16x16x32_bf16(false, a, false, b2, (short)0, acc2, false, false);
        acc3 = __builtin_amdgcn_wmma_f32_16x16x32_bf16(false, a, false, b3, (short)0, acc3, false, false);
    }

    int rbase = m0 + ((lane >> 4) << 3);
    int cl    = lane & 15;
    v8f accs[4] = {acc0, acc1, acc2, acc3};
#pragma unroll
    for (int j = 0; j < 4; j++) {
        int col  = nb + j * 16 + cl;
        float sc = scale ? scale[col] : 1.0f;
        float bi = bias  ? bias[col]  : 0.0f;
#pragma unroll
        for (int e = 0; e < 8; e++) {
            float v  = accs[j][e] * sc + bi;
            if (relu) v = fmaxf(v, 0.0f);
            size_t o = (size_t)(rbase + e) * ldOut + col;
            if (resid) v += resid[o];
            if (outF) outF[o] = v;
            else      outB[o] = (__bf16)v;
        }
    }
    (void)M; (void)N;
}

// ---------------- small prep: fold eval-BN into scale/bias ----------------
__global__ void prep_kernel(const float* conv_b, const float* cbg, const float* cbb,
                            const float* cp_b1, const float* g1, const float* bb1,
                            const float* cp_b2, const float* g2, const float* bb2,
                            float* sConv, float* bConv, float* sC1, float* bC1,
                            float* sC2, float* bC2)
{
    int t = threadIdx.x;                    // 512 threads
    float invs = rsqrtf(1.0f + 1e-5f);
    if (t < 256) { float s = cbg[t] * invs; sConv[t] = s; bConv[t] = conv_b[t] * s + cbb[t]; }
    if (t < 512) { float s = g1[t]  * invs; sC1[t]   = s; bC1[t]   = cp_b1[t]  * s + bb1[t]; }
    if (t < 128) {
        if (t < 113) { float s = g2[t] * invs; sC2[t] = s; bC2[t] = cp_b2[t] * s + bb2[t]; }
        else         { sC2[t] = 0.0f; bC2[t] = 0.0f; }
    }
}

// ---------------- dtype converts ----------------
__global__ void f2b_kernel(const float* __restrict__ s, __bf16* __restrict__ d, int n) {
    int i = blockIdx.x * 256 + threadIdx.x;
    if (i < n) d[i] = (__bf16)s[i];
}
__global__ void f2b_pad_kernel(const float* __restrict__ s, __bf16* __restrict__ d,
                               int rows, int cols, int prows) {
    int i = blockIdx.x * 256 + threadIdx.x;
    if (i >= prows * cols) return;
    int r = i / cols;
    d[i] = (r < rows) ? (__bf16)s[i] : (__bf16)0.0f;
}

// ---------------- stage 1: rect_proj pack / epilogue, recog mean ----------------
__global__ void pack_rect_kernel(const float* __restrict__ rq, __bf16* __restrict__ xp) {
    int i = blockIdx.x * 256 + threadIdx.x;
    if (i >= 24576 * 256) return;
    int c = i & 255;
    int rowp = i >> 8;                      // n*6 + p
    int n = rowp / 6, p = rowp - n * 6;
    xp[i] = (__bf16)rq[(size_t)n * 1536 + c * 6 + p];
}
__global__ void avg6_kernel(const float* __restrict__ H, __bf16* __restrict__ semin) {
    int i = blockIdx.x * 256 + threadIdx.x;
    if (i >= 4096 * 256) return;
    int n = i >> 8, o = i & 255;
    const float* p = H + (size_t)n * 1536 + o;
    float s = 0.0f;
#pragma unroll
    for (int pp = 0; pp < 6; pp++) s += p[pp * 256];
    semin[(size_t)n * 512 + o] = (__bf16)(s * (1.0f / 6.0f));
}
__global__ void recog_mean_kernel(const float* __restrict__ rf, __bf16* __restrict__ rm) {
    int i = blockIdx.x * 256 + threadIdx.x;
    if (i >= 4096 * 256) return;
    int n = i >> 8, d = i & 255;
    const float* p = rf + (size_t)n * 4096 + d;
    float s = 0.0f;
#pragma unroll
    for (int t = 0; t < 16; t++) s += p[t * 256];
    rm[i] = (__bf16)(s * (1.0f / 16.0f));
}

// ---------------- geometry features ----------------
__global__ __launch_bounds__(128)
void feat_kernel(const float* __restrict__ sem, const float* __restrict__ oq,
                 float* __restrict__ fe, float* __restrict__ centers)
{
    int n = blockIdx.x, t = threadIdx.x;
    if (t < 113) fe[(size_t)n * 128 + t] = sem[(size_t)n * 128 + t];
    if (t == 0) {
        float q[8];
#pragma unroll
        for (int i = 0; i < 8; i++) q[i] = oq[(size_t)n * 8 + i] * (1.0f / 1024.0f);
        centers[(size_t)n * 2 + 0] = (q[0] + q[2] + q[4] + q[6]) * 0.25f;
        centers[(size_t)n * 2 + 1] = (q[1] + q[3] + q[5] + q[7]) * 0.25f;
        float h1x = q[6] - q[0], h1y = q[7] - q[1];
        float h2x = q[4] - q[2], h2y = q[5] - q[3];
        float d1x = (q[2] + h2x * 0.5f) - (q[0] + h1x * 0.5f);
        float d1y = (q[3] + h2y * 0.5f) - (q[1] + h1y * 0.5f);
        float w  = sqrtf(d1x * d1x + d1y * d1y);
        float iw = 1.0f / fmaxf(w, 1e-6f);
        float n1x = d1x * iw, n1y = d1y * iw;
        float hx = (h1x + h2x) * 0.5f, hy = (h1y + h2y) * 0.5f;
        float ht = sqrtf(hx * hx + hy * hy);
        float* r = fe + (size_t)n * 128;
#pragma unroll
        for (int i = 0; i < 8; i++) r[113 + i] = q[i];
        r[121] = n1x; r[122] = n1y; r[123] = -n1y; r[124] = n1x;
        r[125] = w;   r[126] = ht;  r[127] = 0.0f;
    }
}

// ---------------- kNN (top-31 by center distance, per region row) ----------------
__global__ __launch_bounds__(256)
void knn_kernel(const float* __restrict__ centers, const float* __restrict__ fe,
                int* __restrict__ idxf, float* __restrict__ darr, float* __restrict__ carr)
{
    int b = blockIdx.x >> 10;
    int f = blockIdx.x & 1023;
    int tid = threadIdx.x;
    int n = (b << 10) + f;
    __shared__ float sd[1024];
    __shared__ float rbm[256];
    __shared__ int   rbi[256];
    __shared__ int   selI[31];
    __shared__ float selD[31];
    __shared__ float geo[4];
    const float* cb = centers + (size_t)(b << 10) * 2;
    float cx = cb[f * 2], cy = cb[f * 2 + 1];
    for (int j = tid; j < 1024; j += 256) {
        float dx = cb[j * 2] - cx, dy = cb[j * 2 + 1] - cy;
        sd[j] = (j == f) ? __builtin_inff() : sqrtf(dx * dx + dy * dy);
    }
    __syncthreads();
    for (int sel = 0; sel < 31; sel++) {
        float bm = __builtin_inff(); int bi = 1 << 30;
        for (int j = tid; j < 1024; j += 256) {
            float d = sd[j];
            if (d < bm) { bm = d; bi = j; }
        }
        rbm[tid] = bm; rbi[tid] = bi;
        __syncthreads();
        for (int s2 = 128; s2 > 0; s2 >>= 1) {
            if (tid < s2) {
                float d2 = rbm[tid + s2]; int i2 = rbi[tid + s2];
                if (d2 < rbm[tid] || (d2 == rbm[tid] && i2 < rbi[tid])) { rbm[tid] = d2; rbi[tid] = i2; }
            }
            __syncthreads();
        }
        if (tid == 0) { selI[sel] = rbi[0]; selD[sel] = rbm[0]; sd[rbi[0]] = __builtin_inff(); }
        __syncthreads();
    }
    if (tid == 0) {
        const float* q = fe + (size_t)n * 128 + 113;
        float p0x = (q[0] + q[6]) * 0.5f, p0y = (q[1] + q[7]) * 0.5f;
        float p1x = (q[2] + q[4]) * 0.5f, p1y = (q[3] + q[5]) * 0.5f;
        float drx = p1x - p0x, dry = p1y - p0y;
        float nr  = fmaxf(sqrtf(drx * drx + dry * dry), 1e-6f);
        geo[0] = drx / nr; geo[1] = dry / nr;
        geo[2] = (p0x + p1x) * 0.5f; geo[3] = (p0y + p1y) * 0.5f;
        idxf[(size_t)n * 32] = 0;
        darr[(size_t)n * 32] = -1.0f;
        carr[(size_t)n * 32] = -2.0f;
    }
    __syncthreads();
    if (tid < 31) {
        int nb2 = selI[tid];
        float vx = cb[nb2 * 2] - geo[2], vy = cb[nb2 * 2 + 1] - geo[3];
        float nr = fmaxf(sqrtf(vx * vx + vy * vy), 1e-6f);
        idxf[(size_t)n * 32 + 1 + tid] = nb2 + 1;
        darr[(size_t)n * 32 + 1 + tid] = selD[tid];
        carr[(size_t)n * 32 + 1 + tid] = (geo[0] * vx + geo[1] * vy) / nr;
    }
}

// ---------------- build transformer input x[131072,256] ----------------
__global__ __launch_bounds__(256)
void buildx_kernel(const float* __restrict__ fe, const int* __restrict__ idxf,
                   const float* __restrict__ darr, const float* __restrict__ carr,
                   float* __restrict__ x)
{
    int row = blockIdx.x;                  // (n,k)
    int c = threadIdx.x;
    int n = row >> 5, k = row & 31, b = n >> 10;
    float v;
    if (c < 127) v = fe[(size_t)n * 128 + c];
    else if (c < 254) {
        if (k == 0) v = 0.0f;
        else {
            int nb = (b << 10) + (idxf[n * 32 + k] - 1);
            v = fe[(size_t)nb * 128 + (c - 127)];
        }
    } else if (c == 254) v = darr[n * 32 + k];
    else                 v = carr[n * 32 + k];
    x[(size_t)row * 256 + c] = v;
}

// ---------------- LayerNorm -> bf16 ----------------
__global__ __launch_bounds__(256)
void ln_kernel(const float* __restrict__ x, const float* __restrict__ g,
               const float* __restrict__ b, __bf16* __restrict__ y, int M)
{
    int row = blockIdx.x * 8 + (threadIdx.x >> 5);
    if (row >= M) return;
    int lane = threadIdx.x & 31;
    const float* xr = x + (size_t)row * 256 + lane * 8;
    float4 p0 = *(const float4*)xr;
    float4 p1 = *(const float4*)(xr + 4);
    float v[8] = {p0.x, p0.y, p0.z, p0.w, p1.x, p1.y, p1.z, p1.w};
    float s = 0.0f, ss = 0.0f;
#pragma unroll
    for (int i = 0; i < 8; i++) { s += v[i]; ss += v[i] * v[i]; }
#pragma unroll
    for (int o = 16; o > 0; o >>= 1) { s += __shfl_xor(s, o, 32); ss += __shfl_xor(ss, o, 32); }
    float mean = s * (1.0f / 256.0f);
    float var  = ss * (1.0f / 256.0f) - mean * mean;
    float rstd = rsqrtf(var + 1e-5f);
    __bf16* yr = y + (size_t)row * 256 + lane * 8;
#pragma unroll
    for (int i = 0; i < 8; i++) {
        int c = lane * 8 + i;
        yr[i] = (__bf16)(((v[i] - mean) * rstd) * g[c] + b[c]);
    }
}

// ---------------- attention: one wave per (sequence, head), seq=32, dh=32 ----------------
__global__ __launch_bounds__(32)
void attn_kernel(const __bf16* __restrict__ qkv, __bf16* __restrict__ o)
{
    int n = blockIdx.x >> 3;
    int h = blockIdx.x & 7;
    int t = threadIdx.x;
    __shared__ float Ks[32][33];
    __shared__ float Vs[32][33];
    const __bf16* base = qkv + (size_t)n * 32 * 768;
    const __bf16* kr = base + (size_t)t * 768 + 256 + h * 32;
    const __bf16* vr = base + (size_t)t * 768 + 512 + h * 32;
#pragma unroll
    for (int d = 0; d < 32; d++) { Ks[t][d] = (float)kr[d]; Vs[t][d] = (float)vr[d]; }
    __syncthreads();
    const __bf16* qr = base + (size_t)t * 768 + h * 32;
    float q[32];
#pragma unroll
    for (int d = 0; d < 32; d++) q[d] = (float)qr[d];
    float s[32];
    float mx = -1e30f;
#pragma unroll
    for (int kt = 0; kt < 32; kt++) {
        float a = 0.0f;
#pragma unroll
        for (int d = 0; d < 32; d++) a += q[d] * Ks[kt][d];
        a *= 0.17677669529663687f;         // 1/sqrt(32)
        s[kt] = a; mx = fmaxf(mx, a);
    }
    float sum = 0.0f;
#pragma unroll
    for (int kt = 0; kt < 32; kt++) { s[kt] = expf(s[kt] - mx); sum += s[kt]; }
    float inv = 1.0f / sum;
    float oa[32] = {};
#pragma unroll
    for (int kt = 0; kt < 32; kt++) {
        float p = s[kt] * inv;
#pragma unroll
        for (int d = 0; d < 32; d++) oa[d] += p * Vs[kt][d];
    }
    __bf16* orow = o + (size_t)(n * 32 + t) * 256 + h * 32;
#pragma unroll
    for (int d = 0; d < 32; d++) orow[d] = (__bf16)oa[d];
}

// ---------------- head: 256 -> 3, wave per row ----------------
__global__ __launch_bounds__(256)
void head_kernel(const float* __restrict__ x, const float* __restrict__ w,
                 const float* __restrict__ hb, float* __restrict__ dots, int M)
{
    int row = blockIdx.x * 8 + (threadIdx.x >> 5);
    if (row >= M) return;
    int lane = threadIdx.x & 31;
    const float* xr = x + (size_t)row * 256 + lane * 8;
    float4 p0 = *(const float4*)xr;
    float4 p1 = *(const float4*)(xr + 4);
    float v[8] = {p0.x, p0.y, p0.z, p0.w, p1.x, p1.y, p1.z, p1.w};
    float s[3];
#pragma unroll
    for (int c = 0; c < 3; c++) {
        const float* wc = w + c * 256 + lane * 8;
        float a = 0.0f;
#pragma unroll
        for (int i = 0; i < 8; i++) a += v[i] * wc[i];
#pragma unroll
        for (int o = 16; o > 0; o >>= 1) a += __shfl_xor(a, o, 32);
        s[c] = a;
    }
    if (lane == 0) {
#pragma unroll
        for (int c = 0; c < 3; c++) dots[(size_t)row * 4 + c] = s[c] + hb[c];
    }
}

// ---------------- output fill / scatter ----------------
__global__ void fill_kernel(float* __restrict__ p, float val, size_t n) {
    size_t i = (size_t)blockIdx.x * blockDim.x + threadIdx.x;
    size_t st = (size_t)gridDim.x * blockDim.x;
    for (; i < n; i += st) p[i] = val;
}
__global__ void scatter_kernel(const float* __restrict__ dots, const int* __restrict__ idxf,
                               float* __restrict__ out) {
    int t = blockIdx.x * 256 + threadIdx.x;
    if (t >= 4096 * 32) return;
    int n = t >> 5;
    int b = n >> 10, f = n & 1023;
    int col = idxf[t];
#pragma unroll
    for (int c = 0; c < 3; c++)
        out[(((size_t)(b * 3 + c) * 1024) + f) * 1025 + col] = dots[(size_t)t * 4 + c];
}

// ---------------- host launch ----------------
extern "C" void kernel_launch(void* const* d_in, const int* in_sizes, int n_in,
                              void* d_out, int out_size, void* d_ws, size_t ws_size,
                              hipStream_t stream)
{
    const float* rq      = (const float*)d_in[0];
    const float* oq      = (const float*)d_in[1];
    const float* rf      = (const float*)d_in[2];
    const float* conv_w  = (const float*)d_in[3];
    const float* conv_b  = (const float*)d_in[4];
    const float* cbg     = (const float*)d_in[5];
    const float* cbb     = (const float*)d_in[6];
    const float* recog_w = (const float*)d_in[7];
    const float* recog_b = (const float*)d_in[8];
    const float* cp_w1   = (const float*)d_in[9];
    const float* cp_b1   = (const float*)d_in[10];
    const float* bn1g    = (const float*)d_in[11];
    const float* bn1b    = (const float*)d_in[12];
    const float* cp_w2   = (const float*)d_in[13];
    const float* cp_b2   = (const float*)d_in[14];
    const float* bn2g    = (const float*)d_in[15];
    const float* bn2b    = (const float*)d_in[16];
    const float* wqkv    = (const float*)d_in[17];
    const float* bqkv    = (const float*)d_in[18];
    const float* wo      = (const float*)d_in[19];
    const float* bo      = (const float*)d_in[20];
    const float* ln1g    = (const float*)d_in[21];
    const float* ln1b    = (const float*)d_in[22];
    const float* ln2g    = (const float*)d_in[23];
    const float* ln2b    = (const float*)d_in[24];
    const float* w1      = (const float*)d_in[25];
    const float* b1      = (const float*)d_in[26];
    const float* w2      = (const float*)d_in[27];
    const float* b2      = (const float*)d_in[28];
    const float* head_w  = (const float*)d_in[29];
    const float* head_b  = (const float*)d_in[30];

    char* ws = (char*)d_ws;
    size_t off = 0;
    auto alloc = [&](size_t bytes) -> char* {
        char* p = ws + off;
        off = (off + bytes + 255) & ~(size_t)255;
        return p;
    };

    __bf16* wConv = (__bf16*)alloc((size_t)65536 * 2);
    __bf16* wRec  = (__bf16*)alloc((size_t)65536 * 2);
    __bf16* wCp1  = (__bf16*)alloc((size_t)262144 * 2);
    __bf16* wCp2  = (__bf16*)alloc((size_t)128 * 512 * 2);
    __bf16* wQkv  = (__bf16*)alloc((size_t)4 * 768 * 256 * 2);
    __bf16* wOp   = (__bf16*)alloc((size_t)4 * 256 * 256 * 2);
    __bf16* wF1   = (__bf16*)alloc((size_t)4 * 512 * 256 * 2);
    __bf16* wF2   = (__bf16*)alloc((size_t)4 * 256 * 512 * 2);
    float* sConv = (float*)alloc(256 * 4);
    float* bConv = (float*)alloc(256 * 4);
    float* sC1   = (float*)alloc(512 * 4);
    float* bC1   = (float*)alloc(512 * 4);
    float* sC2   = (float*)alloc(128 * 4);
    float* bC2   = (float*)alloc(128 * 4);
    __bf16* Xp  = (__bf16*)alloc((size_t)24576 * 256 * 2);
    float*  Hb  = (float*)alloc((size_t)24576 * 256 * 4);
    __bf16* Rm  = (__bf16*)alloc((size_t)4096 * 256 * 2);
    __bf16* Si  = (__bf16*)alloc((size_t)4096 * 512 * 2);
    __bf16* T1  = (__bf16*)alloc((size_t)4096 * 512 * 2);
    float*  Sem = (float*)alloc((size_t)4096 * 128 * 4);
    float*  Fe  = (float*)alloc((size_t)4096 * 128 * 4);
    float*  Ct  = (float*)alloc((size_t)4096 * 2 * 4);
    int*    If  = (int*)alloc((size_t)4096 * 32 * 4);
    float*  Da  = (float*)alloc((size_t)4096 * 32 * 4);
    float*  Ca  = (float*)alloc((size_t)4096 * 32 * 4);
    float*  X   = (float*)alloc((size_t)131072 * 256 * 4);
    __bf16* Y   = (__bf16*)alloc((size_t)131072 * 256 * 2);
    __bf16* Qkv = (__bf16*)alloc((size_t)131072 * 768 * 2);
    __bf16* Ao  = (__bf16*)alloc((size_t)131072 * 256 * 2);
    float*  Dt  = (float*)alloc((size_t)131072 * 4 * 4);
    __bf16* Ff1 = Qkv;   // qkv is dead after attention within a layer

    // BN folding + weight conversion
    prep_kernel<<<1, 512, 0, stream>>>(conv_b, cbg, cbb, cp_b1, bn1g, bn1b,
                                       cp_b2, bn2g, bn2b, sConv, bConv, sC1, bC1, sC2, bC2);
    f2b_kernel<<<(65536 + 255) / 256, 256, 0, stream>>>(conv_w, wConv, 65536);
    f2b_kernel<<<(65536 + 255) / 256, 256, 0, stream>>>(recog_w, wRec, 65536);
    f2b_kernel<<<(262144 + 255) / 256, 256, 0, stream>>>(cp_w1, wCp1, 262144);
    f2b_pad_kernel<<<(128 * 512 + 255) / 256, 256, 0, stream>>>(cp_w2, wCp2, 113, 512, 128);
    f2b_kernel<<<(786432 + 255) / 256, 256, 0, stream>>>(wqkv, wQkv, 786432);
    f2b_kernel<<<(262144 + 255) / 256, 256, 0, stream>>>(wo, wOp, 262144);
    f2b_kernel<<<(524288 + 255) / 256, 256, 0, stream>>>(w1, wF1, 524288);
    f2b_kernel<<<(524288 + 255) / 256, 256, 0, stream>>>(w2, wF2, 524288);

    // stage 1: rect_proj (conv1x1 + BN + ReLU -> avg) and recog encoder
    pack_rect_kernel<<<(24576 * 256 + 255) / 256, 256, 0, stream>>>(rq, Xp);
    gemm_bf16<<<dim3(4, 192), 256, 0, stream>>>(Xp, wConv, sConv, bConv, nullptr,
                                                Hb, nullptr, 24576, 256, 256, 256, 1);
    avg6_kernel<<<4096, 256, 0, stream>>>(Hb, Si);
    recog_mean_kernel<<<4096, 256, 0, stream>>>(rf, Rm);
    gemm_bf16<<<dim3(4, 32), 256, 0, stream>>>(Rm, wRec, nullptr, recog_b, nullptr,
                                               nullptr, Si + 256, 4096, 256, 256, 512, 0);
    // content projection (512->512 BN ReLU, 512->113pad BN ReLU)
    gemm_bf16<<<dim3(8, 32), 256, 0, stream>>>(Si, wCp1, sC1, bC1, nullptr,
                                               nullptr, T1, 4096, 512, 512, 512, 1);
    gemm_bf16<<<dim3(2, 32), 256, 0, stream>>>(T1, wCp2, sC2, bC2, nullptr,
                                               Sem, nullptr, 4096, 128, 512, 128, 1);
    // geometry + kNN + transformer input
    feat_kernel<<<4096, 128, 0, stream>>>(Sem, oq, Fe, Ct);
    knn_kernel<<<4096, 256, 0, stream>>>(Ct, Fe, If, Da, Ca);
    buildx_kernel<<<131072, 256, 0, stream>>>(Fe, If, Da, Ca, X);

    // 4-layer pre-norm transformer
    for (int l = 0; l < 4; l++) {
        ln_kernel<<<16384, 256, 0, stream>>>(X, ln1g + l * 256, ln1b + l * 256, Y, 131072);
        gemm_bf16<<<dim3(12, 1024), 256, 0, stream>>>(Y, wQkv + (size_t)l * 196608, nullptr,
                                                      bqkv + l * 768, nullptr, nullptr, Qkv,
                                                      131072, 768, 256, 768, 0);
        attn_kernel<<<32768, 32, 0, stream>>>(Qkv, Ao);
        gemm_bf16<<<dim3(4, 1024), 256, 0, stream>>>(Ao, wOp + (size_t)l * 65536, nullptr,
                                                     bo + l * 256, X, X, nullptr,
                                                     131072, 256, 256, 256, 0);
        ln_kernel<<<16384, 256, 0, stream>>>(X, ln2g + l * 256, ln2b + l * 256, Y, 131072);
        gemm_bf16<<<dim3(8, 1024), 256, 0, stream>>>(Y, wF1 + (size_t)l * 131072, nullptr,
                                                     b1 + l * 512, nullptr, nullptr, Ff1,
                                                     131072, 512, 256, 512, 1);
        gemm_bf16<<<dim3(4, 1024), 256, 0, stream>>>(Ff1, wF2 + (size_t)l * 131072, nullptr,
                                                     b2 + l * 256, X, X, nullptr,
                                                     131072, 256, 512, 256, 0);
    }

    // head + -inf scatter output
    head_kernel<<<16384, 256, 0, stream>>>(X, head_w, head_b, Dt, 131072);
    fill_kernel<<<2048, 256, 0, stream>>>((float*)d_out, -INFINITY, (size_t)out_size);
    scatter_kernel<<<(4096 * 32 + 255) / 256, 256, 0, stream>>>(Dt, If, (float*)d_out);

    (void)in_sizes; (void)n_in; (void)ws_size;
}
